// LEOCEM_41601053229216
// MI455X (gfx1250) — matrix operand order repaired and checked by reading, not compiled, not run on hardware
//
#include <hip/hip_runtime.h>

// ---------------------------------------------------------------------------
// CEM energy-model sampler for MI455X (gfx1250, wave32).
// One workgroup per batch row; all 10 CEM iterations run locally out of LDS.
// Middle layer (128x128) runs on v_wmma_f32_16x16x32_f16.
// ---------------------------------------------------------------------------

typedef __attribute__((ext_vector_type(16))) _Float16 v16h;
typedef __attribute__((ext_vector_type(8)))  _Float16 v8h;
typedef __attribute__((ext_vector_type(8)))  float    v8f;

#define NB       2048
#define CEM_NS   1000
#define CEM_NE   100
#define CEM_NI   10
#define NS_FINAL 1000
#define HDIM     128
#define NTILES   63          // ceil(1000/16)
#define THREADS  256
#define NWAVES   8

// LDS layout (bytes). Total 76188 -> launch with 76800 dynamic LDS.
// 4 blocks/WGP fit under the 320KB budget.
#define OFF_W2T  0            // f16 [n][k]  128*128*2 = 32768
#define OFF_H1   32768        // f16 [wave][m][k] 8*16*128*2 = 32768
#define OFF_W1X  65536        // f32 [128]
#define OFF_W1Y  66048
#define OFF_B1   66560
#define OFF_B2   67072
#define OFF_W3   67584
#define OFF_FX   68096        // f32 [1008]
#define OFF_YS   72128        // f32 [1008]
#define OFF_SC   76160        // f32 mu, sigma, sumY, sumY2
#define OFF_CNT  76176        // u32 cnt, cntLess, tieCnt
#define LDS_BYTES 76800

__device__ __forceinline__ unsigned pcg(unsigned v) {
  v = v * 747796405u + 2891336453u;
  unsigned w = ((v >> ((v >> 28u) + 4u)) ^ v) * 277803737u;
  return (w >> 22u) ^ w;
}
__device__ __forceinline__ float u01(unsigned h) {
  return ((float)(h >> 8) + 0.5f) * 5.9604645e-8f;   // (0,1)
}
__device__ __forceinline__ float gauss(unsigned s) {
  float u1 = u01(pcg(s));
  float u2 = u01(pcg(s ^ 0x9E3779B9u));
  return sqrtf(fmaxf(-2.0f * __logf(u1), 0.0f)) * __cosf(6.28318530718f * u2);
}
__device__ __forceinline__ float tanh_fast(float x) {
  x = fminf(fmaxf(x, -15.0f), 15.0f);
  float e = __expf(x + x);                           // v_exp_f32
  return (e - 1.0f) / (e + 1.0f);
}

__global__ void __launch_bounds__(THREADS, 1)
leocem_kernel(const float* __restrict__ gx,
              const float* __restrict__ gW1, const float* __restrict__ gb1,
              const float* __restrict__ gW2, const float* __restrict__ gb2,
              const float* __restrict__ gW3, const float* __restrict__ gb3,
              float* __restrict__ out)
{
  extern __shared__ __attribute__((aligned(128))) char smem[];
  _Float16* sW2T = (_Float16*)(smem + OFF_W2T);   // W2 transposed, f16
  _Float16* sH1  = (_Float16*)(smem + OFF_H1);    // per-wave h1 tile slabs
  float* sW1x = (float*)(smem + OFF_W1X);
  float* sW1y = (float*)(smem + OFF_W1Y);
  float* sB1  = (float*)(smem + OFF_B1);
  float* sB2  = (float*)(smem + OFF_B2);
  float* sW3  = (float*)(smem + OFF_W3);
  float* sFX  = (float*)(smem + OFF_FX);
  float* sYS  = (float*)(smem + OFF_YS);
  float* sSc  = (float*)(smem + OFF_SC);          // mu, sigma, sumY, sumY2
  unsigned* sCnt = (unsigned*)(smem + OFF_CNT);   // cnt, cntLess, tieCnt

  const int tid   = threadIdx.x;
  const int wave  = tid >> 5;
  const int lane  = tid & 31;
  const int mrow  = lane & 15;    // M (candidate in tile) / N (feature in tile)
  const int half  = lane >> 4;
  const int batch = blockIdx.x;

  // ---- stage weights into LDS (once per block; weights stay hot in L2) ----
  for (int i = tid; i < HDIM; i += THREADS) {
    sW1x[i] = gW1[i];
    sW1y[i] = gW1[HDIM + i];
    sB1[i]  = gb1[i];
    sB2[i]  = gb2[i];
    sW3[i]  = gW3[i];
  }
  for (int i = tid; i < HDIM * HDIM; i += THREADS) {
    const int k = i >> 7, n = i & 127;
    sW2T[n * HDIM + k] = (_Float16)gW2[i];        // transpose so B-frags are contiguous
  }
  if (tid == 0) { sSc[0] = 0.0f; sSc[1] = 10.0f; }  // mu0, INIT_SIGMA
  const float xval = gx[batch];
  const float b3   = gb3[0];
  __syncthreads();

  _Float16* slabRow = sH1 + wave * (16 * HDIM) + mrow * HDIM;

  for (int iter = 0; iter < CEM_NI; ++iter) {
    const float mu = sSc[0], sigma = sSc[1];

    // ---------------- candidate evaluation (per-wave tiles) ----------------
    for (int t = wave; t < NTILES; t += NWAVES) {
      const int c0   = t * 16;
      const int cand = c0 + mrow;

      // candidate sample y = mu + sigma * N(0,1), deterministic per (batch,iter,cand)
      const unsigned seed =
          ((unsigned)batch * (unsigned)CEM_NI + (unsigned)iter) * 1024u + (unsigned)cand;
      const float y = mu + sigma * gauss(seed);
      if (half == 0) sYS[cand] = y;

      // layer 1 (VALU): lane computes features [half*64, half*64+64) of its row
      #pragma unroll 8
      for (int f = 0; f < 64; ++f) {
        const int ff = half * 64 + f;
        const float h = tanh_fast(fmaf(xval, sW1x[ff], fmaf(y, sW1y[ff], sB1[ff])));
        slabRow[ff] = (_Float16)h;
      }
      // intra-wave LDS RAW (partner lane wrote the other half of our row)
      asm volatile("s_wait_dscnt 0" ::: "memory");

      // A fragments: 16x32 f16 per ISA layout
      //   lanes 0-15 : M=lane,   K = kk*32 + {0..7, 16..23}
      //   lanes 16-31: M=lane-16,K = kk*32 + {8..15, 24..31}
      v16h afrag[4];
      #pragma unroll
      for (int kk = 0; kk < 4; ++kk) {
        const int k0 = kk * 32 + half * 8;
        const v8h lo = *(const v8h*)(slabRow + k0);
        const v8h hi = *(const v8h*)(slabRow + k0 + 16);
        #pragma unroll
        for (int e = 0; e < 8; ++e) { afrag[kk][e] = lo[e]; afrag[kk][8 + e] = hi[e]; }
      }

      float ep[8];
      #pragma unroll
      for (int i = 0; i < 8; ++i) ep[i] = 0.0f;

      // layer 2: 8 N-tiles x 4 K-steps of v_wmma_f32_16x16x32_f16
      #pragma unroll
      for (int nt = 0; nt < 8; ++nt) {
        v8f acc = {};
        const _Float16* brow = sW2T + (nt * 16 + mrow) * HDIM;  // B: lane = N
        #pragma unroll
        for (int kk = 0; kk < 4; ++kk) {
          // B 32x16 layout: lanes 0-15 K={0..7,16..23}, lanes 16-31 K={8..15,24..31}
          const v8h blo = *(const v8h*)(brow + kk * 32 + half * 8);
          const v8h bhi = *(const v8h*)(brow + kk * 32 + half * 8 + 16);
          v16h b;
          #pragma unroll
          for (int e = 0; e < 8; ++e) { b[e] = blo[e]; b[8 + e] = bhi[e]; }
          acc = __builtin_amdgcn_wmma_f32_16x16x32_f16(
              false, afrag[kk], false, b, (short)0, acc, false, false);
        }
        // fused epilogue: bias + tanh + W3 dot on the C tile
        // C layout: VGPR i -> M = i + 8*half, lane&15 -> N
        const int   n   = nt * 16 + mrow;
        const float w3n = sW3[n], b2n = sB2[n];
        #pragma unroll
        for (int i = 0; i < 8; ++i)
          ep[i] = fmaf(tanh_fast(acc[i] + b2n), w3n, ep[i]);
      }

      // reduce each row's partial energy over the 16 lanes of its half-wave
      #pragma unroll
      for (int i = 0; i < 8; ++i) {
        float v = ep[i];
        v += __shfl_xor(v, 1, 32);
        v += __shfl_xor(v, 2, 32);
        v += __shfl_xor(v, 4, 32);
        v += __shfl_xor(v, 8, 32);
        if (mrow == 0) {
          const float E = v + b3;
          sFX[c0 + i + 8 * half] = E * E;
        }
      }
    }
    __syncthreads();

    // ------------- top-100: radix bisection on float bits (fX >= 0) --------
    unsigned blo = 0u, bhi = 0xFFFFFFFFu;
    for (int it = 0; it < 32; ++it) {
      const unsigned mid = blo + ((bhi - blo) >> 1);
      if (tid == 0) sCnt[0] = 0u;
      __syncthreads();
      unsigned c = 0;
      for (int s = tid; s < CEM_NS; s += THREADS)
        c += (__float_as_uint(sFX[s]) <= mid) ? 1u : 0u;
      if (c) atomicAdd(&sCnt[0], c);
      __syncthreads();
      const unsigned total = sCnt[0];
      if (total >= CEM_NE) bhi = mid; else blo = mid + 1u;
      __syncthreads();
    }
    const unsigned T = blo;   // value of the 100th smallest

    if (tid == 0) { sSc[2] = 0.0f; sSc[3] = 0.0f; sCnt[1] = 0u; sCnt[2] = 0u; }
    __syncthreads();
    for (int s = tid; s < CEM_NS; s += THREADS) {
      if (__float_as_uint(sFX[s]) < T) {
        const float yv = sYS[s];
        atomicAdd(&sSc[2], yv);
        atomicAdd(&sSc[3], yv * yv);
        atomicAdd(&sCnt[1], 1u);
      }
    }
    __syncthreads();
    const unsigned less = sCnt[1];
    for (int s = tid; s < CEM_NS; s += THREADS) {
      if (__float_as_uint(sFX[s]) == T) {
        const unsigned slot = atomicAdd(&sCnt[2], 1u);
        if (less + slot < CEM_NE) {        // fill ties to exactly 100 elites
          const float yv = sYS[s];
          atomicAdd(&sSc[2], yv);
          atomicAdd(&sSc[3], yv * yv);
        }
      }
    }
    __syncthreads();
    if (tid == 0) {
      const float nm    = 1.0f / (float)CEM_NE;
      const float newMu = sSc[2] * nm;
      const float var   = fmaxf(sSc[3] * nm - newMu * newMu, 0.0f);
      sSc[0] = newMu;
      sSc[1] = sqrtf(var);
    }
    __syncthreads();
  }

  // ------------------------------ outputs ----------------------------------
  const float muF = sSc[0];
  float var = sSc[1] * sSc[1];
  var = fminf(fmaxf(var, 1e-3f), 100.0f);    // clip(ycov/TEMP, MIN_COV, MAX_COV)
  const float sd = sqrtf(var);
  if (tid == 0) out[batch] = muF;            // yhat
  for (int s = tid; s < NS_FINAL; s += THREADS) {
    const unsigned seed = 0x5bd1e995u ^ ((unsigned)batch * 1024u + (unsigned)s + 0x10000000u);
    out[NB + (size_t)s * NB + batch] = fmaf(sd, gauss(seed), muF);  // ysamples
  }
}

extern "C" void kernel_launch(void* const* d_in, const int* in_sizes, int n_in,
                              void* d_out, int out_size, void* d_ws, size_t ws_size,
                              hipStream_t stream) {
  (void)in_sizes; (void)n_in; (void)out_size; (void)d_ws; (void)ws_size;
  const float* x  = (const float*)d_in[0];
  const float* W1 = (const float*)d_in[1];
  const float* b1 = (const float*)d_in[2];
  const float* W2 = (const float*)d_in[3];
  const float* b2 = (const float*)d_in[4];
  const float* W3 = (const float*)d_in[5];
  const float* b3 = (const float*)d_in[6];
  float* out = (float*)d_out;

  leocem_kernel<<<dim3(NB), dim3(THREADS), LDS_BYTES, stream>>>(
      x, W1, b1, W2, b2, W3, b3, out);
}
